// CostumeQuantizer_90709709291570
// MI455X (gfx1250) — matrix operand order repaired and checked by reading, not compiled, not run on hardware
//
#include <hip/hip_runtime.h>
#include <hip/hip_bf16.h>
#include <math.h>

// ---------------------------------------------------------------------------
// Residual VQ codec for MI455X (gfx1250, wave32, WMMA).
// All matmuls -> v_wmma_f32_16x16x32_f16 (f16 inputs, f32 accumulate).
// dist = ||r||^2 - 2 r.e + ||e||^2 ; r2/e2 kept in exact f32.
// commit_s = mean(new_residual^2); quantized = h - r_final (no q tensors).
//
// Stage kernel: codebook B tiles staged block-wide into double-buffered LDS
// (4x L2 traffic cut, loads for tile bt+1 in flight under the 8 back-to-back
// WMMAs of tile bt).
// ---------------------------------------------------------------------------

#define TOKENS 32768   // B*T = 8*4096
#define DDIM   512
#define DC     256
#define NQ     8
#define BINS   1024

typedef __attribute__((ext_vector_type(16))) _Float16 v16h;
typedef __attribute__((ext_vector_type(8)))  _Float16 v8h;
typedef __attribute__((ext_vector_type(8)))  float    v8f;
typedef __attribute__((ext_vector_type(4)))  float    v4f;

// A-operand (16x32 f16) from an LDS-resident token row, contiguous k order, same as the B loads:
// lane L (m = L&15) element i -> K = (L>>4)*16 + i
__device__ __forceinline__ v16h load_a16(const _Float16* row, int kbase, int khalf) {
    v16h a = *(const v16h*)(row + kbase + khalf * 16);
    return a;
}

__device__ __forceinline__ v8h cvt8(v4f a, v4f b) {
    v8h o;
    o[0] = (_Float16)a[0]; o[1] = (_Float16)a[1];
    o[2] = (_Float16)a[2]; o[3] = (_Float16)a[3];
    o[4] = (_Float16)b[0]; o[5] = (_Float16)b[1];
    o[6] = (_Float16)b[2]; o[7] = (_Float16)b[3];
    return o;
}

__device__ __forceinline__ v8f wmma_f16(v16h a, v16h b, v8f c) {
    return __builtin_amdgcn_wmma_f32_16x16x32_f16(false, a, false, b,
                                                  (short)0, c, false, false);
}

// ---------------------------------------------------------------------------
// Prep: convert in_w / out_w to f16, zero commit accumulators
// ---------------------------------------------------------------------------
__global__ __launch_bounds__(256) void k_prep_weights(
        const float* __restrict__ in_w, const float* __restrict__ out_w,
        _Float16* __restrict__ inw16, _Float16* __restrict__ outw16,
        float* __restrict__ commit_acc) {
    int i = blockIdx.x * 256 + threadIdx.x;            // 0 .. 2*512*256-1
    if (i < DDIM * DC) inw16[i] = (_Float16)in_w[i];
    else               outw16[i - DDIM * DC] = (_Float16)out_w[i - DDIM * DC];
    if (i < NQ) commit_acc[i] = 0.0f;
}

// Prep: per-bin ||e||^2 + f16 codebook conversion. One block per (q,bin).
__global__ __launch_bounds__(256) void k_prep_codebook(
        const float* __restrict__ cb, _Float16* __restrict__ cb16,
        float* __restrict__ e2) {
    __shared__ float red[256];
    int b = blockIdx.x;              // q*1024 + bin
    int t = threadIdx.x;             // Dc = 256 = blockDim
    float v = cb[(size_t)b * DC + t];
    cb16[(size_t)b * DC + t] = (_Float16)v;
    red[t] = v * v;
    __syncthreads();
#pragma unroll
    for (int st = 128; st > 0; st >>= 1) {
        if (t < st) red[t] += red[t + st];
        __syncthreads();
    }
    if (t == 0) e2[b] = red[0];
}

// ---------------------------------------------------------------------------
// Input projection: h = tanh(x @ in_w^T + in_b).  One 16-token tile per wave.
// ---------------------------------------------------------------------------
__global__ __launch_bounds__(128) void k_proj_in(
        const float* __restrict__ x, const _Float16* __restrict__ inw16,
        const float* __restrict__ in_b,
        float* __restrict__ h, float* __restrict__ r) {
    __shared__ _Float16 xt[4][16][DDIM];               // 64 KB
    const int lane = threadIdx.x & 31, w = threadIdx.x >> 5;
    const int tbase = (blockIdx.x * 4 + w) * 16;
    const int m = lane & 15, half = lane >> 4;

    // Stage x tile as f16: lane covers token m, 256 contiguous dims.
    {
        const float* src = x + (size_t)(tbase + m) * DDIM + half * 256;
        _Float16* dst = &xt[w][m][half * 256];
#pragma unroll 4
        for (int i = 0; i < 256; i += 8) {
            v4f a = *(const v4f*)(src + i);
            v4f b = *(const v4f*)(src + i + 4);
            *(v8h*)(dst + i) = cvt8(a, b);
        }
    }
    __syncthreads();

    const _Float16* arow = &xt[w][m][0];
    for (int nt = 0; nt < DC / 16; ++nt) {
        const int n = nt * 16 + m;                     // output channel (column)
        const float bias = in_b[n];
        v8f acc;
#pragma unroll
        for (int i = 0; i < 8; ++i) acc[i] = bias;
        const _Float16* brow = inw16 + (size_t)n * DDIM + half * 16;
        if (nt + 1 < DC / 16)                          // -> global_prefetch_b8
            __builtin_prefetch(brow + (size_t)16 * DDIM, 0, 1);
        // one-ahead pipelined B: wmma kt waits only on the load from kt-1
        v16h B0 = *(const v16h*)(brow);
#pragma unroll
        for (int kt = 0; kt < DDIM / 32 - 1; ++kt) {
            v16h B1 = *(const v16h*)(brow + (kt + 1) * 32);
            acc = wmma_f16(load_a16(arow, kt * 32, half), B0, acc);
            B0 = B1;
        }
        acc = wmma_f16(load_a16(arow, DDIM - 32, half), B0, acc);

        const int rowbase = tbase + half * 8;          // D-layout rows
#pragma unroll
        for (int v = 0; v < 8; ++v) {
            float hv = tanhf(acc[v]);                  // CODEC_RANGE == 1.0
            size_t off = (size_t)(rowbase + v) * DC + n;
            h[off] = hv;
            r[off] = hv;
        }
    }
}

// ---------------------------------------------------------------------------
// One RVQ stage: block-wide double-buffered LDS staging of codebook tiles,
// 8 back-to-back WMMAs per tile, cross-lane argmin, f32 residual update.
// ---------------------------------------------------------------------------
__global__ __launch_bounds__(128) void k_stage(
        const float* __restrict__ cb,        // original f32 codebooks [NQ][BINS][DC]
        const _Float16* __restrict__ cb16,   // f16 copy
        const float* __restrict__ e2,        // [NQ][BINS]
        float* __restrict__ r,               // residual, updated in place
        int* __restrict__ idx_out,           // [NQ][TOKENS]
        float* __restrict__ commit_acc,      // [NQ]
        int s) {
    __shared__ _Float16 rt[4][16][DC];                 // 32 KB : per-wave A tiles
    __shared__ _Float16 bb[2][16][DC];                 // 16 KB : B double buffer
    __shared__ float p2[4][32];
    __shared__ float r2s[4][16];
    __shared__ int idxl[4][16];
    const int tid = threadIdx.x;
    const int lane = tid & 31, w = tid >> 5;
    const int tbase = (blockIdx.x * 4 + w) * 16;
    const int m = lane & 15, half = lane >> 4;

    // ---- stage residual tile as f16; exact f32 ||r||^2 partials ----
    float part = 0.0f;
    {
        const float* src = r + (size_t)(tbase + m) * DC + half * 128;
        _Float16* dst = &rt[w][m][half * 128];
#pragma unroll 4
        for (int i = 0; i < 128; i += 8) {
            v4f a = *(const v4f*)(src + i);
            v4f b = *(const v4f*)(src + i + 4);
            part += a[0]*a[0] + a[1]*a[1] + a[2]*a[2] + a[3]*a[3]
                  + b[0]*b[0] + b[1]*b[1] + b[2]*b[2] + b[3]*b[3];
            *(v8h*)(dst + i) = cvt8(a, b);
        }
    }
    p2[w][lane] = part;
    __syncthreads();
    if (lane < 16) r2s[w][lane] = p2[w][lane] + p2[w][lane + 16];
    __syncthreads();
    float r2v[8];
#pragma unroll
    for (int v = 0; v < 8; ++v) r2v[v] = r2s[w][half * 8 + v];

    // Hoist the wave's A tile into registers (8 k-tiles, 64 VGPRs).
    const _Float16* arow = &rt[w][m][0];
    v16h Areg[8];
#pragma unroll
    for (int kt = 0; kt < 8; ++kt) Areg[kt] = load_a16(arow, kt * 32, half);

    const _Float16* cbs16 = cb16 + (size_t)s * BINS * DC;
    const float*    e2s   = e2 + s * BINS;

    // This thread's share of each 8 KB B tile: row tid>>3, 32 halves at (tid&7)*32
    const int brow_s = tid >> 3;
    const int bcol_s = (tid & 7) * 32;
    const _Float16* gsrc = cbs16 + (size_t)brow_s * DC + bcol_s;

    // Prologue: stage tile 0 into bb[0].
    {
        v8h t0 = *(const v8h*)(gsrc + 0);
        v8h t1 = *(const v8h*)(gsrc + 8);
        v8h t2 = *(const v8h*)(gsrc + 16);
        v8h t3 = *(const v8h*)(gsrc + 24);
        _Float16* d = &bb[0][brow_s][bcol_s];
        *(v8h*)(d + 0) = t0;  *(v8h*)(d + 8)  = t1;
        *(v8h*)(d + 16) = t2; *(v8h*)(d + 24) = t3;
    }
    __syncthreads();

    float bestD[8]; int bestI[8];
#pragma unroll
    for (int v = 0; v < 8; ++v) { bestD[v] = 3.4e38f; bestI[v] = 0; }

    for (int bt = 0; bt < BINS / 16 - 1; ++bt) {
        const int cur = bt & 1, nxt = cur ^ 1;
        // Issue global loads for tile bt+1 (in flight during the WMMA block).
        const _Float16* g = gsrc + (size_t)(bt + 1) * 16 * DC;
        v8h t0 = *(const v8h*)(g + 0);
        v8h t1 = *(const v8h*)(g + 8);
        v8h t2 = *(const v8h*)(g + 16);
        v8h t3 = *(const v8h*)(g + 24);

        // 8 back-to-back WMMAs on tile bt out of LDS.
        const int n = bt * 16 + m;
        const float e2n = e2s[n];
        const _Float16* lrow = &bb[cur][m][half * 16];
        v8f acc = {};
#pragma unroll
        for (int kt = 0; kt < DC / 32; ++kt) {
            v16h B = *(const v16h*)(lrow + kt * 32);
            acc = wmma_f16(Areg[kt], B, acc);
        }
#pragma unroll
        for (int v = 0; v < 8; ++v) {
            float d = r2v[v] - 2.0f * acc[v] + e2n;
            if (d < bestD[v] || (d == bestD[v] && n < bestI[v])) {
                bestD[v] = d; bestI[v] = n;
            }
        }
        // Retire the staged loads into the other LDS buffer.
        _Float16* d = &bb[nxt][brow_s][bcol_s];
        *(v8h*)(d + 0) = t0;  *(v8h*)(d + 8)  = t1;
        *(v8h*)(d + 16) = t2; *(v8h*)(d + 24) = t3;
        __syncthreads();
    }
    {   // Peeled last tile (bt = 63, buffer 1).
        const int bt = BINS / 16 - 1;
        const int n = bt * 16 + m;
        const float e2n = e2s[n];
        const _Float16* lrow = &bb[bt & 1][m][half * 16];
        v8f acc = {};
#pragma unroll
        for (int kt = 0; kt < DC / 32; ++kt) {
            v16h B = *(const v16h*)(lrow + kt * 32);
            acc = wmma_f16(Areg[kt], B, acc);
        }
#pragma unroll
        for (int v = 0; v < 8; ++v) {
            float d = r2v[v] - 2.0f * acc[v] + e2n;
            if (d < bestD[v] || (d == bestD[v] && n < bestI[v])) {
                bestD[v] = d; bestI[v] = n;
            }
        }
    }

    // Reduce argmin across the 16 lanes sharing each 8-token group.
#pragma unroll
    for (int v = 0; v < 8; ++v) {
        float d = bestD[v]; int i = bestI[v];
#pragma unroll
        for (int msk = 1; msk < 16; msk <<= 1) {
            float od = __shfl_xor(d, msk);
            int   oi = __shfl_xor(i, msk);
            if (od < d || (od == d && oi < i)) { d = od; i = oi; }
        }
        bestD[v] = d; bestI[v] = i;
    }
    if (m == 0) {
#pragma unroll
        for (int v = 0; v < 8; ++v) {
            int tok = tbase + half * 8 + v;
            idx_out[(size_t)s * TOKENS + tok] = bestI[v];
            idxl[w][half * 8 + v] = bestI[v];
        }
    }
    __syncthreads();

    // Residual update in exact f32; commit_s = sum(new_r^2).
    const float* cbs = cb + (size_t)s * BINS * DC;
    float sq = 0.0f;
    for (int t = 0; t < 16; ++t) {
        const float* cv  = cbs + (size_t)idxl[w][t] * DC;
        float*       row = r + (size_t)(tbase + t) * DC;
#pragma unroll
        for (int j = lane; j < DC; j += 32) {
            float nv = row[j] - cv[j];
            row[j] = nv;
            sq += nv * nv;
        }
    }
#pragma unroll
    for (int msk = 1; msk < 32; msk <<= 1) sq += __shfl_xor(sq, msk);
    if (lane == 0) atomicAdd(&commit_acc[s], sq);
}

// ---------------------------------------------------------------------------
// Output projection: out = (h - r_final) @ out_w^T + out_b
// ---------------------------------------------------------------------------
__global__ __launch_bounds__(128) void k_proj_out(
        const float* __restrict__ h, const float* __restrict__ r,
        const _Float16* __restrict__ outw16, const float* __restrict__ out_b,
        float* __restrict__ out) {
    __shared__ _Float16 qt[4][16][DC];                 // 32 KB
    const int lane = threadIdx.x & 31, w = threadIdx.x >> 5;
    const int tbase = (blockIdx.x * 4 + w) * 16;
    const int m = lane & 15, half = lane >> 4;

    {
        const float* hs = h + (size_t)(tbase + m) * DC + half * 128;
        const float* rs = r + (size_t)(tbase + m) * DC + half * 128;
        _Float16* dst = &qt[w][m][half * 128];
#pragma unroll 4
        for (int i = 0; i < 128; i += 8) {
            v4f ha = *(const v4f*)(hs + i);
            v4f hb = *(const v4f*)(hs + i + 4);
            v4f ra = *(const v4f*)(rs + i);
            v4f rb = *(const v4f*)(rs + i + 4);
            *(v8h*)(dst + i) = cvt8(ha - ra, hb - rb);
        }
    }
    __syncthreads();

    const _Float16* arow = &qt[w][m][0];
    for (int nt = 0; nt < DDIM / 16; ++nt) {
        const int n = nt * 16 + m;
        const float bias = out_b[n];
        v8f acc;
#pragma unroll
        for (int i = 0; i < 8; ++i) acc[i] = bias;
        const _Float16* brow = outw16 + (size_t)n * DC + half * 16;
        // one-ahead pipelined B
        v16h B0 = *(const v16h*)(brow);
#pragma unroll
        for (int kt = 0; kt < DC / 32 - 1; ++kt) {
            v16h B1 = *(const v16h*)(brow + (kt + 1) * 32);
            acc = wmma_f16(load_a16(arow, kt * 32, half), B0, acc);
            B0 = B1;
        }
        acc = wmma_f16(load_a16(arow, DC - 32, half), B0, acc);

        const int rowbase = tbase + half * 8;
#pragma unroll
        for (int v = 0; v < 8; ++v)
            out[(size_t)(rowbase + v) * DDIM + n] = acc[v];
    }
}

__global__ void k_finalize(const float* __restrict__ commit_acc,
                           float* __restrict__ out_commit) {
    if (threadIdx.x == 0 && blockIdx.x == 0) {
        float s = 0.0f;
        for (int i = 0; i < NQ; ++i) s += commit_acc[i];
        *out_commit = 0.1f * (s / (float)(TOKENS * DC));
    }
}

// ---------------------------------------------------------------------------
extern "C" void kernel_launch(void* const* d_in, const int* in_sizes, int n_in,
                              void* d_out, int out_size, void* d_ws, size_t ws_size,
                              hipStream_t stream) {
    const float* x    = (const float*)d_in[0];   // [8,4096,512]
    const float* in_w = (const float*)d_in[1];   // [256,512]
    const float* in_b = (const float*)d_in[2];   // [256]
    const float* outw = (const float*)d_in[3];   // [512,256]
    const float* outb = (const float*)d_in[4];   // [512]
    const float* cb   = (const float*)d_in[5];   // [8,1024,256]

    // d_out: out[TOKENS*D] f32, idxs[NQ*TOKENS] i32 (bit-stored), commit[1]
    float* out        = (float*)d_out;
    int*   idx_out    = (int*)((float*)d_out + (size_t)TOKENS * DDIM);
    float* out_commit = (float*)d_out + (size_t)TOKENS * DDIM + (size_t)NQ * TOKENS;

    // Workspace carve-up (all offsets 256B-aligned; ~72 MB total)
    float*    h      = (float*)d_ws;                       // 32 MB
    float*    r      = h + (size_t)TOKENS * DC;            // 32 MB
    _Float16* cb16   = (_Float16*)(r + (size_t)TOKENS * DC); // 4 MB
    float*    e2     = (float*)(cb16 + (size_t)NQ * BINS * DC); // 32 KB
    _Float16* inw16  = (_Float16*)(e2 + NQ * BINS);        // 256 KB
    _Float16* outw16 = inw16 + (size_t)DDIM * DC;          // 256 KB
    float*    cacc   = (float*)(outw16 + (size_t)DDIM * DC);

    // 1. prep
    k_prep_weights<<<2 * DDIM * DC / 256, 256, 0, stream>>>(in_w, outw, inw16, outw16, cacc);
    k_prep_codebook<<<NQ * BINS, 256, 0, stream>>>(cb, cb16, e2);

    // 2. input projection (WMMA f16): 512 blocks x 4 waves x 16 tokens
    k_proj_in<<<TOKENS / 64, 128, 0, stream>>>(x, inw16, in_b, h, r);

    // 3. eight RVQ stages (WMMA scoring + argmin + residual update)
    for (int s = 0; s < NQ; ++s)
        k_stage<<<TOKENS / 64, 128, 0, stream>>>(cb, cb16, e2, r, idx_out, cacc, s);

    // 4. output projection (WMMA f16)
    k_proj_out<<<TOKENS / 64, 128, 0, stream>>>(h, r, outw16, outb, out);

    // 5. commit loss scalar
    k_finalize<<<1, 1, 0, stream>>>(cacc, out_commit);
}
